// HGTModel_27307402068192
// MI455X (gfx1250) — compile-verified, matrix operand order checked by reading, not run on hardware
//
#include <hip/hip_runtime.h>

#define THREADS 256

typedef __attribute__((ext_vector_type(16))) __bf16 v16bf;
typedef __attribute__((ext_vector_type(8)))  __bf16 v8bf;
typedef __attribute__((ext_vector_type(8)))  float  v8f;

__device__ __forceinline__ unsigned short f2bf(float f) {
  unsigned u = __float_as_uint(f);
  unsigned r = 0x7FFFu + ((u >> 16) & 1u);
  return (unsigned short)((u + r) >> 16);
}

__device__ __forceinline__ v16bf load_frag(const unsigned short* p) {
  v8bf lo = *reinterpret_cast<const v8bf*>(p);
  v8bf hi = *reinterpret_cast<const v8bf*>(p + 16);
  return __builtin_shufflevector(lo, hi,
      0, 1, 2, 3, 4, 5, 6, 7, 8, 9, 10, 11, 12, 13, 14, 15);
}

// ---------------------------------------------------------------------------
// WMMA bf16 GEMM: C[M x ldc] (+bias) = A16[M x lda] * Bt16[rows=N x ldb]^T
//   A row-major bf16; Bt row n = column n of B (B^T row-major).
//   Tile counts MT (row tiles) and NT (col tiles) are compile-time so the
//   inner loop is straight-line WMMA with full EXEC (no saveexec masking).
//   One wave -> (MT*16) x (NT*16) output tile.
// ---------------------------------------------------------------------------
template<int MT, int NT>
__global__ void __launch_bounds__(THREADS)
wmma_gemm_bf16_t(const unsigned short* __restrict__ A, int lda,
                 const unsigned short* __restrict__ B, int ldb,
                 float* __restrict__ C, unsigned short* __restrict__ C16, int ldc,
                 const float* __restrict__ bias,
                 int mtiles, int ngroups, int nbase0, int Kp)
{
  const int wid  = (blockIdx.x * THREADS + threadIdx.x) >> 5;
  const int lane = threadIdx.x & 31;
  if (wid >= mtiles * ngroups) return;
  const int mt    = wid % mtiles;
  const int ng    = wid / mtiles;
  const int nbase = nbase0 + ng * (NT * 16);

  const int kb = (lane < 16) ? 0 : 8;   // per-lane K sub-run base (A/B 16-bit layout)
  const unsigned short* aRow[MT];
  const unsigned short* bRow[NT];
  #pragma unroll
  for (int m = 0; m < MT; ++m)
    aRow[m] = A + (size_t)((mt * MT + m) * 16 + (lane & 15)) * lda + kb;
  #pragma unroll
  for (int t = 0; t < NT; ++t)
    bRow[t] = B + (size_t)(nbase + t * 16 + (lane & 15)) * ldb + kb;

  const v8f z = {0.f, 0.f, 0.f, 0.f, 0.f, 0.f, 0.f, 0.f};
  v8f acc[MT][NT];
  #pragma unroll
  for (int m = 0; m < MT; ++m)
    #pragma unroll
    for (int t = 0; t < NT; ++t) acc[m][t] = z;

  for (int k0 = 0; k0 < Kp; k0 += 32) {
    v16bf afrag[MT];
    #pragma unroll
    for (int m = 0; m < MT; ++m) {
      afrag[m] = load_frag(aRow[m] + k0);
      if (k0 + 32 < Kp) __builtin_prefetch((const void*)(aRow[m] + k0 + 32), 0, 1);
    }
    #pragma unroll
    for (int t = 0; t < NT; ++t) {
      v16bf bfrag = load_frag(bRow[t] + k0);
      #pragma unroll
      for (int m = 0; m < MT; ++m) {
        acc[m][t] = __builtin_amdgcn_wmma_f32_16x16x32_bf16(
            false, afrag[m], false, bfrag, (short)0, acc[m][t], false, false);
      }
    }
  }

  const int rofs = (lane >> 4) << 3;   // acc VGPR i -> row i (lanes 0-15) or 8+i
  #pragma unroll
  for (int m = 0; m < MT; ++m) {
    #pragma unroll
    for (int t = 0; t < NT; ++t) {
      const int col = nbase + t * 16 + (lane & 15);
      const float bb = bias ? bias[col] : 0.f;
      #pragma unroll
      for (int i = 0; i < 8; ++i) {
        const int row = (mt * MT + m) * 16 + rofs + i;
        float v = acc[m][t][i] + bb;
        if (C)   C[(size_t)row * ldc + col]   = v;
        if (C16) C16[(size_t)row * ldc + col] = f2bf(v);
      }
    }
  }
}

// --------------------------- elementwise kernels ---------------------------
__global__ void conv_f2bf_kernel(const float* __restrict__ x,
                                 unsigned short* __restrict__ y, long long n) {
  long long i = (long long)blockIdx.x * blockDim.x + threadIdx.x;
  if (i < n) y[i] = f2bf(x[i]);
}
__global__ void relu_f2bf_kernel(const float* __restrict__ x,
                                 unsigned short* __restrict__ y, long long n) {
  long long i = (long long)blockIdx.x * blockDim.x + threadIdx.x;
  if (i < n) { float v = x[i]; y[i] = f2bf(v > 0.f ? v : 0.f); }
}
__global__ void gelu_f2bf_kernel(const float* __restrict__ x,
                                 unsigned short* __restrict__ y, long long n) {
  long long i = (long long)blockIdx.x * blockDim.x + threadIdx.x;
  if (i < n) {
    float v = x[i];
    y[i] = f2bf(0.5f * v * (1.f + erff(v * 0.70710678118654752f)));
  }
}
__global__ void fill_f32_kernel(float* p, float v, long long n) {
  long long i = (long long)blockIdx.x * blockDim.x + threadIdx.x;
  if (i < n) p[i] = v;
}
__global__ void fill_u32_kernel(unsigned* p, unsigned v, long long n) {
  long long i = (long long)blockIdx.x * blockDim.x + threadIdx.x;
  if (i < n) p[i] = v;
}

// ---------------------- weight conversion (pad + transpose) ----------------
// W [Kin x 2D] row-major, b [2D] -> Wt [2*Dp x Kp] bf16 (B^T layout), bt [2*Dp]
__global__ void conv_w_qkv_kernel(const float* __restrict__ W, const float* __restrict__ b,
                                  unsigned short* __restrict__ Wt, float* __restrict__ bt,
                                  int Kin, int Kp, int D, int Dp) {
  int idx = blockIdx.x * blockDim.x + threadIdx.x;
  int HDp = 2 * Dp;
  if (idx >= HDp * Kp) return;
  int n = idx / Kp, kk = idx % Kp;
  int h = n / Dp, d = n % Dp;
  int out = 2 * D;
  float v = (d < D && kk < Kin) ? W[(size_t)kk * out + h * D + d] : 0.f;
  Wt[(size_t)n * Kp + kk] = f2bf(v);
  if (kk == 0) bt[n] = (d < D) ? b[h * D + d] : 0.f;
}
// Wa [2D x Nout] -> Wat [NoutPad x 2*Dp] bf16 (row n = out-column n; K = (h,d) padded)
__global__ void conv_w_a_kernel(const float* __restrict__ W, const float* __restrict__ b,
                                unsigned short* __restrict__ Wt, float* __restrict__ bt,
                                int D, int Dp, int Nout, int NoutPad) {
  int idx = blockIdx.x * blockDim.x + threadIdx.x;
  int HDp = 2 * Dp;
  if (idx >= NoutPad * HDp) return;
  int n = idx / HDp, c = idx % HDp;
  int h = c / Dp, d = c % Dp;
  float v = (n < Nout && d < D) ? W[(size_t)(h * D + d) * Nout + n] : 0.f;
  Wt[(size_t)n * HDp + c] = f2bf(v);
  if (c == 0) bt[n] = (n < Nout) ? b[n] : 0.f;
}
// A [2 x D x D] -> At [2 x Dp x Dp] bf16 with At[h][n][kk] = A[h][kk][n]
__global__ void conv_rel_kernel(const float* __restrict__ A,
                                unsigned short* __restrict__ At, int D, int Dp) {
  int idx = blockIdx.x * blockDim.x + threadIdx.x;
  if (idx >= 2 * Dp * Dp) return;
  int h = idx / (Dp * Dp);
  int rem = idx % (Dp * Dp);
  int n = rem / Dp, kk = rem % Dp;
  float v = (n < D && kk < D) ? A[((size_t)h * D + kk) * D + n] : 0.f;
  At[idx] = f2bf(v);
}

// ------------------------------ edge kernels -------------------------------
// One wave per (edge, head): logits[e,h] = <q[dst],k_rel[src]> * p_rel[h]/sqrt(D)
__global__ void __launch_bounds__(THREADS)
logits_kernel(const float* __restrict__ q, const float* __restrict__ krel,
              const int* __restrict__ src, const int* __restrict__ dst,
              const float* __restrict__ prel, int Dreal, int Dp,
              float* __restrict__ out, int E) {
  int wid  = (blockIdx.x * THREADS + threadIdx.x) >> 5;
  int lane = threadIdx.x & 31;
  if (wid >= E * 2) return;
  int e = wid >> 1, h = wid & 1;
  const float* qp = q    + (size_t)dst[e] * 2 * Dp + h * Dp;
  const float* kp = krel + (size_t)src[e] * 2 * Dp + h * Dp;
  float acc = 0.f;
  for (int i = lane; i < Dp; i += 32) acc += qp[i] * kp[i];
  for (int o = 16; o > 0; o >>= 1) acc += __shfl_xor(acc, o, 32);
  if (lane == 0) out[e * 2 + h] = acc * prel[h] * rsqrtf((float)Dreal);
}
// order-preserving float->uint encoding for atomic segment max
__global__ void segmax_kernel(const float* __restrict__ lg, const int* __restrict__ dst,
                              unsigned* __restrict__ m, int E) {
  int i = blockIdx.x * blockDim.x + threadIdx.x;
  if (i >= E * 2) return;
  int e = i >> 1, h = i & 1;
  unsigned b = __float_as_uint(lg[i]);
  unsigned enc = (b & 0x80000000u) ? ~b : (b | 0x80000000u);
  atomicMax(&m[dst[e] * 2 + h], enc);
}
__global__ void expsum_kernel(float* __restrict__ lg, const int* __restrict__ dst,
                              const unsigned* __restrict__ m, float* __restrict__ s, int E) {
  int i = blockIdx.x * blockDim.x + threadIdx.x;
  if (i >= E * 2) return;
  int e = i >> 1, h = i & 1;
  unsigned em = m[dst[e] * 2 + h];
  float mv = (em & 0x80000000u) ? __uint_as_float(em & 0x7FFFFFFFu)
                                : __uint_as_float(~em);
  float el = expf(lg[i] - mv);
  lg[i] = el;
  atomicAdd(&s[dst[e] * 2 + h], el);
}
// One wave per (edge, head): agg[dst] += v_rel[src] * att
__global__ void __launch_bounds__(THREADS)
agg_kernel(const float* __restrict__ vrel, const float* __restrict__ att,
           const float* __restrict__ s, const int* __restrict__ src,
           const int* __restrict__ dst, float* __restrict__ agg, int Dp, int E) {
  int wid  = (blockIdx.x * THREADS + threadIdx.x) >> 5;
  int lane = threadIdx.x & 31;
  if (wid >= E * 2) return;
  int e = wid >> 1, h = wid & 1;
  int si = src[e], di = dst[e];
  float w = att[e * 2 + h] / (s[di * 2 + h] + 1e-16f);
  const float* vp = vrel + (size_t)si * 2 * Dp + h * Dp;
  float*       ap = agg  + (size_t)di * 2 * Dp + h * Dp;
  for (int i = lane; i < Dp; i += 32) atomicAdd(&ap[i], vp[i] * w);
}

__global__ void copy_trim_kernel(const float* __restrict__ src, float* __restrict__ dst,
                                 int N, int lds, int W) {
  long long i = (long long)blockIdx.x * blockDim.x + threadIdx.x;
  long long total = (long long)N * W;
  if (i >= total) return;
  long long n = i / W, c = i % W;
  dst[i] = src[(size_t)n * lds + c];
}

// ------------------------------- host side ---------------------------------
static inline dim3 g1(long long n) { return dim3((unsigned)((n + THREADS - 1) / THREADS)); }
static inline dim3 gw(long long waves) { return dim3((unsigned)((waves + 7) / 8)); }

// Dispatch full 64-col groups with NT=4 and a column tail with NT=1/2/3.
// MT=2 when M % 32 == 0 (paper: 100000), else MT=1 (author: 50000).
template<int MT>
static void launch_gemm_mt(const unsigned short* A, int lda,
                           const unsigned short* B, int ldb,
                           float* C, unsigned short* C16, int ldc,
                           const float* bias, int M, int Npad, int Kp,
                           hipStream_t st) {
  const int mtiles = M / (MT * 16);
  const int fullgr = Npad >> 6;
  const int remNT  = (Npad & 63) >> 4;
  if (fullgr > 0) {
    long long w = (long long)mtiles * fullgr;
    wmma_gemm_bf16_t<MT, 4><<<gw(w), THREADS, 0, st>>>(
        A, lda, B, ldb, C, C16, ldc, bias, mtiles, fullgr, 0, Kp);
  }
  if (remNT == 1)
    wmma_gemm_bf16_t<MT, 1><<<gw(mtiles), THREADS, 0, st>>>(
        A, lda, B, ldb, C, C16, ldc, bias, mtiles, 1, fullgr << 6, Kp);
  else if (remNT == 2)
    wmma_gemm_bf16_t<MT, 2><<<gw(mtiles), THREADS, 0, st>>>(
        A, lda, B, ldb, C, C16, ldc, bias, mtiles, 1, fullgr << 6, Kp);
  else if (remNT == 3)
    wmma_gemm_bf16_t<MT, 3><<<gw(mtiles), THREADS, 0, st>>>(
        A, lda, B, ldb, C, C16, ldc, bias, mtiles, 1, fullgr << 6, Kp);
}

static inline void launch_gemm(const unsigned short* A, int lda,
                               const unsigned short* B, int ldb,
                               float* C, unsigned short* C16, int ldc,
                               const float* bias, int M, int Npad, int Kp,
                               hipStream_t st) {
  if ((M & 31) == 0) launch_gemm_mt<2>(A, lda, B, ldb, C, C16, ldc, bias, M, Npad, Kp, st);
  else               launch_gemm_mt<1>(A, lda, B, ldb, C, C16, ldc, bias, M, Npad, Kp, st);
}

extern "C" void kernel_launch(void* const* d_in, const int* in_sizes, int n_in,
                              void* d_out, int out_size, void* d_ws, size_t ws_size,
                              hipStream_t stream) {
  (void)n_in; (void)out_size; (void)ws_size;
  const int IN = 128;
  const int NP = in_sizes[0] / IN;   // papers  (100000)
  const int NA = in_sizes[1] / IN;   // authors (50000)
  const int E  = in_sizes[56] / 2;   // 250000 edges per relation
  const int Nt[2] = {NA, NP};        // type 0 = author, 1 = paper

  const float* xin[2] = {(const float*)d_in[1], (const float*)d_in[0]};
  // param leaf order per layer (pytree sorted keys), relations sorted: cites, writes, written_by
  // 0 Wa_a 1 Wa_p 2 Wk_a 3 Wk_p 4 Wq_a 5 Wq_p 6 Wv_a 7 Wv_p 8..10 a_rel
  // 11 ba_a 12 ba_p 13 bk_a 14 bk_p 15 bq_a 16 bq_p 17 bv_a 18 bv_p
  // 19..21 m_rel 22..24 p_rel 25 skip_a 26 skip_p
  const int* ei[3] = {(const int*)d_in[58], (const int*)d_in[56], (const int*)d_in[57]};
  const int relsrc[3] = {1, 0, 1};   // cites: p->p, writes: a->p, written_by: p->a
  const int reldst[3] = {1, 1, 0};

  // workspace bump allocator
  char* wsb = (char*)d_ws; size_t off = 0;
  auto alloc = [&](size_t bytes) -> void* {
    void* p = wsb + off; off = (off + bytes + 255) & ~(size_t)255; return p;
  };
  const int HDpMax = 320, KpMax = 256, NoutPadMax = 304, DpMax = 160;

  unsigned short *wkt[2], *wqt[2], *wvt[2], *wat[2];
  float *bkt[2], *bqt[2], *bvt[2], *bat[2];
  unsigned short *xin16[2], *k16[2], *v16[2], *g16[2];
  float *qb[2], *aggb[2], *ob[2];
  for (int t = 0; t < 2; ++t) {
    wkt[t] = (unsigned short*)alloc((size_t)HDpMax * KpMax * 2);
    wqt[t] = (unsigned short*)alloc((size_t)HDpMax * KpMax * 2);
    wvt[t] = (unsigned short*)alloc((size_t)HDpMax * KpMax * 2);
    wat[t] = (unsigned short*)alloc((size_t)NoutPadMax * HDpMax * 2);
    bkt[t] = (float*)alloc((size_t)HDpMax * 4);
    bqt[t] = (float*)alloc((size_t)HDpMax * 4);
    bvt[t] = (float*)alloc((size_t)HDpMax * 4);
    bat[t] = (float*)alloc((size_t)NoutPadMax * 4);
    xin16[t] = (unsigned short*)alloc((size_t)Nt[t] * KpMax * 2);
    k16[t]   = (unsigned short*)alloc((size_t)Nt[t] * HDpMax * 2);
    v16[t]   = (unsigned short*)alloc((size_t)Nt[t] * HDpMax * 2);
    g16[t]   = (unsigned short*)alloc((size_t)Nt[t] * HDpMax * 2);
    qb[t]    = (float*)alloc((size_t)Nt[t] * HDpMax * 4);
    aggb[t]  = (float*)alloc((size_t)Nt[t] * HDpMax * 4);
    ob[t]    = (float*)alloc((size_t)Nt[t] * NoutPadMax * 4);
  }
  unsigned short* relAT = (unsigned short*)alloc((size_t)2 * DpMax * DpMax * 2);
  unsigned short* relMT = (unsigned short*)alloc((size_t)2 * DpMax * DpMax * 2);
  float* krel = (float*)alloc((size_t)NP * HDpMax * 4);
  float* vrel = (float*)alloc((size_t)NP * HDpMax * 4);
  float* att  = (float*)alloc((size_t)E * 2 * 4);
  float* sbuf = (float*)alloc((size_t)NP * 2 * 4);
  unsigned* mbuf = (unsigned*)alloc((size_t)NP * 2 * 4);

  struct LD { int Kin, Kp, D, Dp, Nout, NoutPad; };
  const LD Ld[2] = { {128, 128, 128, 128, 256, 256},
                     {256, 256, 150, 160, 300, 304} };

  // initial inputs -> bf16
  conv_f2bf_kernel<<<g1((long long)NA * IN), THREADS, 0, stream>>>(xin[0], xin16[0], (long long)NA * IN);
  conv_f2bf_kernel<<<g1((long long)NP * IN), THREADS, 0, stream>>>(xin[1], xin16[1], (long long)NP * IN);

  for (int layer = 0; layer < 2; ++layer) {
    const LD& L = Ld[layer];
    const int HDp = 2 * L.Dp;
    const int base = 2 + layer * 27;

    for (int t = 0; t < 2; ++t) {
      long long nq = (long long)HDp * L.Kp;
      conv_w_qkv_kernel<<<g1(nq), THREADS, 0, stream>>>(
          (const float*)d_in[base + 2 + t], (const float*)d_in[base + 13 + t],
          wkt[t], bkt[t], L.Kin, L.Kp, L.D, L.Dp);
      conv_w_qkv_kernel<<<g1(nq), THREADS, 0, stream>>>(
          (const float*)d_in[base + 4 + t], (const float*)d_in[base + 15 + t],
          wqt[t], bqt[t], L.Kin, L.Kp, L.D, L.Dp);
      conv_w_qkv_kernel<<<g1(nq), THREADS, 0, stream>>>(
          (const float*)d_in[base + 6 + t], (const float*)d_in[base + 17 + t],
          wvt[t], bvt[t], L.Kin, L.Kp, L.D, L.Dp);
      long long na = (long long)L.NoutPad * HDp;
      conv_w_a_kernel<<<g1(na), THREADS, 0, stream>>>(
          (const float*)d_in[base + 0 + t], (const float*)d_in[base + 11 + t],
          wat[t], bat[t], L.D, L.Dp, L.Nout, L.NoutPad);
    }

    for (int t = 0; t < 2; ++t) {
      launch_gemm(xin16[t], L.Kp, wqt[t], L.Kp, qb[t], nullptr, HDp, bqt[t], Nt[t], HDp, L.Kp, stream);
      launch_gemm(xin16[t], L.Kp, wkt[t], L.Kp, nullptr, k16[t], HDp, bkt[t], Nt[t], HDp, L.Kp, stream);
      launch_gemm(xin16[t], L.Kp, wvt[t], L.Kp, nullptr, v16[t], HDp, bvt[t], Nt[t], HDp, L.Kp, stream);
      fill_f32_kernel<<<g1((long long)Nt[t] * HDp), THREADS, 0, stream>>>(
          aggb[t], 0.f, (long long)Nt[t] * HDp);
    }

    for (int r = 0; r < 3; ++r) {
      const int ts = relsrc[r], td = reldst[r];
      conv_rel_kernel<<<g1(2LL * L.Dp * L.Dp), THREADS, 0, stream>>>(
          (const float*)d_in[base + 8 + r], relAT, L.D, L.Dp);
      conv_rel_kernel<<<g1(2LL * L.Dp * L.Dp), THREADS, 0, stream>>>(
          (const float*)d_in[base + 19 + r], relMT, L.D, L.Dp);
      for (int h = 0; h < 2; ++h) {
        launch_gemm(k16[ts] + h * L.Dp, HDp, relAT + (size_t)h * L.Dp * L.Dp, L.Dp,
                    krel + h * L.Dp, nullptr, HDp, nullptr, Nt[ts], L.Dp, L.Dp, stream);
        launch_gemm(v16[ts] + h * L.Dp, HDp, relMT + (size_t)h * L.Dp * L.Dp, L.Dp,
                    vrel + h * L.Dp, nullptr, HDp, nullptr, Nt[ts], L.Dp, L.Dp, stream);
      }
      fill_u32_kernel<<<g1((long long)Nt[td] * 2), THREADS, 0, stream>>>(mbuf, 0u, (long long)Nt[td] * 2);
      fill_f32_kernel<<<g1((long long)Nt[td] * 2), THREADS, 0, stream>>>(sbuf, 0.f, (long long)Nt[td] * 2);

      const int* srcI = ei[r];
      const int* dstI = ei[r] + E;
      long long nw = (long long)E * 2;
      logits_kernel<<<gw(nw), THREADS, 0, stream>>>(
          qb[td], krel, srcI, dstI, (const float*)d_in[base + 22 + r], L.D, L.Dp, att, E);
      segmax_kernel<<<g1(nw), THREADS, 0, stream>>>(att, dstI, mbuf, E);
      expsum_kernel<<<g1(nw), THREADS, 0, stream>>>(att, dstI, mbuf, sbuf, E);
      agg_kernel<<<gw(nw), THREADS, 0, stream>>>(vrel, att, sbuf, srcI, dstI, aggb[td], L.Dp, E);
    }

    for (int t = 0; t < 2; ++t) {
      long long n = (long long)Nt[t] * HDp;
      gelu_f2bf_kernel<<<g1(n), THREADS, 0, stream>>>(aggb[t], g16[t], n);
      launch_gemm(g16[t], HDp, wat[t], HDp, ob[t], nullptr, L.NoutPad, bat[t],
                  Nt[t], L.NoutPad, HDp, stream);
    }

    if (layer == 0) {
      for (int t = 0; t < 2; ++t) {
        long long n = (long long)Nt[t] * 256;   // L1 out (256) == L2 Kp (256), exact
        relu_f2bf_kernel<<<g1(n), THREADS, 0, stream>>>(ob[t], xin16[t], n);
      }
    }
  }

  float* out = (float*)d_out;
  copy_trim_kernel<<<g1((long long)NP * 300), THREADS, 0, stream>>>(ob[1], out, NP, 304, 300);
  copy_trim_kernel<<<g1((long long)NA * 300), THREADS, 0, stream>>>(ob[0], out + (size_t)NP * 300, NA, 304, 300);
}
